// Model_11441792876961
// MI455X (gfx1250) — compile-verified, matrix operand order checked by reading, not run on hardware
//
#include <hip/hip_runtime.h>
#include <math.h>

#define NN 100000
#define EE 1600000
#define DIN 128
#define DOUT 256
#define KORD 10

typedef __attribute__((ext_vector_type(2))) float v2f;
typedef __attribute__((ext_vector_type(8))) float v8f;

#if defined(__has_builtin)
#if __has_builtin(__builtin_amdgcn_wmma_f32_16x16x4_f32)
#define HAVE_WMMA_F32 1
#endif
#endif

// ---------------- preprocessing ----------------

__global__ void edge_counts_kernel(const int* __restrict__ row, const int* __restrict__ col,
                                   float* __restrict__ deg, int* __restrict__ cnt) {
  int e = blockIdx.x * blockDim.x + threadIdx.x;
  if (e >= EE) return;
  atomicAdd(&deg[row[e]], 1.0f);
  atomicAdd(&cnt[col[e]], 1);
}

__global__ void dinv_kernel(const float* __restrict__ deg, float* __restrict__ dinv) {
  int i = blockIdx.x * blockDim.x + threadIdx.x;
  if (i >= NN) return;
  float d = deg[i];
  dinv[i] = (d > 0.f) ? rsqrtf(fmaxf(d, 1e-12f)) : 0.f;
}

#define SCAN_CHUNK 1024
#define SCAN_NBLK ((NN + SCAN_CHUNK - 1) / SCAN_CHUNK)

__global__ void scan_pass1(const int* __restrict__ cnt, int* __restrict__ blksum) {
  __shared__ int sh[256];
  int blk = blockIdx.x, t = threadIdx.x;
  int base = blk * SCAN_CHUNK + t * 4;
  int s = 0;
#pragma unroll
  for (int k = 0; k < 4; ++k) { int idx = base + k; if (idx < NN) s += cnt[idx]; }
  sh[t] = s; __syncthreads();
  for (int off = 128; off > 0; off >>= 1) {
    if (t < off) sh[t] += sh[t + off];
    __syncthreads();
  }
  if (t == 0) blksum[blk] = sh[0];
}

__global__ void scan_pass2(int* __restrict__ blksum, int* __restrict__ csr_ptr) {
  if (threadIdx.x == 0 && blockIdx.x == 0) {
    int run = 0;
    for (int i = 0; i < SCAN_NBLK; ++i) { int v = blksum[i]; blksum[i] = run; run += v; }
    csr_ptr[NN] = run;
  }
}

__global__ void scan_pass3(const int* __restrict__ cnt, const int* __restrict__ blksum,
                           int* __restrict__ csr_ptr, int* __restrict__ fill) {
  __shared__ int sh[256];
  int blk = blockIdx.x, t = threadIdx.x;
  int base = blk * SCAN_CHUNK + t * 4;
  int loc[4]; int s = 0;
#pragma unroll
  for (int k = 0; k < 4; ++k) {
    int idx = base + k;
    int v = (idx < NN) ? cnt[idx] : 0;
    loc[k] = s; s += v;
  }
  sh[t] = s; __syncthreads();
  for (int off = 1; off < 256; off <<= 1) {
    int v = (t >= off) ? sh[t - off] : 0;
    __syncthreads();
    sh[t] += v;
    __syncthreads();
  }
  int excl = sh[t] - s + blksum[blk];
#pragma unroll
  for (int k = 0; k < 4; ++k) {
    int idx = base + k;
    if (idx < NN) { csr_ptr[idx] = excl + loc[k]; fill[idx] = excl + loc[k]; }
  }
}

__global__ void scatter_kernel(const int* __restrict__ row, const int* __restrict__ col,
                               const float* __restrict__ dinv, int* __restrict__ fill,
                               int* __restrict__ csr_src, float* __restrict__ csr_w) {
  int e = blockIdx.x * blockDim.x + threadIdx.x;
  if (e >= EE) return;
  int r = row[e], c = col[e];
  int pos = atomicAdd(&fill[c], 1);
  csr_src[pos] = r;
  csr_w[pos] = -dinv[r] * dinv[c];
}

// coe[0..10] = highpass coefficients, coe[16..26] = lowpass
__global__ void coe_kernel(const float* __restrict__ temp, float* __restrict__ coe) {
  int i = threadIdx.x;
  if (i > KORD) return;
  const float piok = 3.14159265358979323846f / (KORD + 1);
  float shp = 0.f, slp = 0.f;
  for (int j = 0; j <= KORD; ++j) {
    float m = cosf((float)i * ((float)(KORD - j) + 0.5f) * piok);
    shp += m * fmaxf(temp[j], 0.f);
    slp += m * fmaxf(temp[KORD - j], 0.f);
  }
  coe[i] = (2.f / (KORD + 1)) * shp;
  coe[16 + i] = (2.f / (KORD + 1)) * slp;
}

// ---------------- Chebyshev propagation ----------------

__global__ void init_acc_kernel(const float* __restrict__ x, float* __restrict__ ahp,
                                float* __restrict__ alp, const float* __restrict__ coe) {
  long i = (long)blockIdx.x * blockDim.x + threadIdx.x;
  const long total = (long)NN * DIN / 4;
  if (i >= total) return;
  float chp = coe[0] * 0.5f, clp = coe[16] * 0.5f;
  float4 xv = reinterpret_cast<const float4*>(x)[i];
  float4 h, l;
  h.x = chp * xv.x; h.y = chp * xv.y; h.z = chp * xv.z; h.w = chp * xv.w;
  l.x = clp * xv.x; l.y = clp * xv.y; l.z = clp * xv.z; l.w = clp * xv.w;
  reinterpret_cast<float4*>(ahp)[i] = h;
  reinterpret_cast<float4*>(alp)[i] = l;
}

// one wave32 per destination node; lane handles 4 contiguous features (float4)
// t = A*prop(xin) + B*tx0;  txout = t;  ahp += coe_hp[ci]*t;  alp += coe_lp[ci]*t
__global__ void __launch_bounds__(256) prop_step_kernel(
    const float* __restrict__ xin, const float* __restrict__ tx0,
    float* __restrict__ txout, float* __restrict__ ahp, float* __restrict__ alp,
    const int* __restrict__ csr_ptr, const int* __restrict__ csr_src,
    const float* __restrict__ csr_w, const float* __restrict__ coe,
    int ci, float A, float B) {
  int wid = (blockIdx.x * blockDim.x + threadIdx.x) >> 5;
  int lane = threadIdx.x & 31;
  if (wid >= NN) return;
  int beg = csr_ptr[wid], end = csr_ptr[wid + 1];
  int fo = lane * 4;
  float ax = 0.f, ay = 0.f, az = 0.f, aw = 0.f;
  for (int e = beg; e < end; ++e) {
    int s = csr_src[e];
    float w = csr_w[e];
    float4 xv = *reinterpret_cast<const float4*>(xin + (long)s * DIN + fo);
    ax = fmaf(w, xv.x, ax); ay = fmaf(w, xv.y, ay);
    az = fmaf(w, xv.z, az); aw = fmaf(w, xv.w, aw);
  }
  long base = (long)wid * DIN + fo;
  float tx = A * ax, ty = A * ay, tz = A * az, tw = A * aw;
  if (B != 0.f) {
    float4 t0 = *reinterpret_cast<const float4*>(tx0 + base);
    tx = fmaf(B, t0.x, tx); ty = fmaf(B, t0.y, ty);
    tz = fmaf(B, t0.z, tz); tw = fmaf(B, t0.w, tw);
  }
  float4 t; t.x = tx; t.y = ty; t.z = tz; t.w = tw;
  *reinterpret_cast<float4*>(txout + base) = t;
  float chp = coe[ci], clp = coe[16 + ci];
  float4 h = *reinterpret_cast<const float4*>(ahp + base);
  h.x = fmaf(chp, tx, h.x); h.y = fmaf(chp, ty, h.y);
  h.z = fmaf(chp, tz, h.z); h.w = fmaf(chp, tw, h.w);
  *reinterpret_cast<float4*>(ahp + base) = h;
  float4 l = *reinterpret_cast<const float4*>(alp + base);
  l.x = fmaf(clp, tx, l.x); l.y = fmaf(clp, ty, l.y);
  l.z = fmaf(clp, tz, l.z); l.w = fmaf(clp, tw, l.w);
  *reinterpret_cast<float4*>(alp + base) = l;
}

// ---------------- WMMA GEMM (16x16 tile over K=128) ----------------

__device__ __forceinline__ v8f gemm_tile16(const float* __restrict__ Ar,
                                           const float* __restrict__ Wg,
                                           int n0, int lane) {
  const int half = lane >> 4, lr = lane & 15;
  v8f c = {0.f, 0.f, 0.f, 0.f, 0.f, 0.f, 0.f, 0.f};
#if HAVE_WMMA_F32
  const float* a_base = Ar + lr * DIN;
#pragma unroll
  for (int kk = 0; kk < DIN; kk += 4) {
    int k = kk + half * 2;
    float2 af = *reinterpret_cast<const float2*>(a_base + k);
    v2f a; a.x = af.x; a.y = af.y;
    v2f b; b.x = Wg[(long)k * DOUT + n0 + lr];
    b.y = Wg[(long)(k + 1) * DOUT + n0 + lr];
    c = __builtin_amdgcn_wmma_f32_16x16x4_f32(false, a, false, b, (short)0, c, false, false);
  }
#else
  typedef __attribute__((ext_vector_type(16))) _Float16 v16h;
#pragma unroll
  for (int kk = 0; kk < DIN; kk += 32) {
    v16h a, b;
#pragma unroll
    for (int e = 0; e < 16; ++e) {
      int k = kk + ((e < 8) ? (8 * half + e) : (16 + 8 * half + (e - 8)));
      a[e] = (_Float16)Ar[lr * DIN + k];
      b[e] = (_Float16)Wg[(long)k * DOUT + n0 + lr];
    }
    c = __builtin_amdgcn_wmma_f32_16x16x32_f16(false, a, false, b, (short)0, c, false, false);
  }
#endif
  return c;
}

// h = relu(acc@W + b); colsum[n] += scale * sum_rows h  (one wave per 16x16 tile)
__global__ void __launch_bounds__(256) gemm_colsum_kernel(
    const float* __restrict__ acc, const float* __restrict__ Wg,
    const float* __restrict__ bias, const float* __restrict__ scale,
    float* __restrict__ colsum) {
  int wid = (blockIdx.x * blockDim.x + threadIdx.x) >> 5;
  int lane = threadIdx.x & 31;
  const int ntiles = (NN / 16) * (DOUT / 16);
  if (wid >= ntiles) return;
  int mt = wid >> 4, nt = wid & 15;
  int m0 = mt * 16, n0 = nt * 16;
  v8f c = gemm_tile16(acc + (long)m0 * DIN, Wg, n0, lane);
  int lr = lane & 15;
  float bn = bias[n0 + lr];
  float s = 0.f;
#pragma unroll
  for (int r = 0; r < 8; ++r) s += fmaxf(c[r] + bn, 0.f);
  s += __shfl_xor(s, 16, 32);           // combine the two row-halves of the tile
  if (lane < 16) atomicAdd(&colsum[n0 + lane], scale[0] * s);
}

// c = relu(colsum/N); v = bil_w @ c
__global__ void make_v_kernel(const float* __restrict__ colsum, const float* __restrict__ bilw,
                              float* __restrict__ v) {
  __shared__ float c[DOUT];
  int t = threadIdx.x;
  c[t] = fmaxf(colsum[t] * (1.0f / (float)NN), 0.f);
  __syncthreads();
  float s = 0.f;
  for (int j = 0; j < DOUT; ++j) s = fmaf(bilw[(long)t * DOUT + j], c[j], s);
  v[t] = s;
}

// out[m] = relu(acc@W + b)[m,:] @ v + bilb   (one wave per 16-row block, loops all n-tiles)
__global__ void __launch_bounds__(256) gemm_score_kernel(
    const float* __restrict__ acc, const float* __restrict__ Wg,
    const float* __restrict__ bias, const float* __restrict__ v,
    const float* __restrict__ bilb, float* __restrict__ outseg) {
  int wid = (blockIdx.x * blockDim.x + threadIdx.x) >> 5;
  int lane = threadIdx.x & 31;
  if (wid >= NN / 16) return;
  int m0 = wid * 16;
  int half = lane >> 4, lr = lane & 15;
  const float* Ar = acc + (long)m0 * DIN;
  float sc[8];
#pragma unroll
  for (int r = 0; r < 8; ++r) sc[r] = 0.f;
  for (int nt = 0; nt < DOUT / 16; ++nt) {
    int n0 = nt * 16;
    v8f c = gemm_tile16(Ar, Wg, n0, lane);
    float bn = bias[n0 + lr];
    float vn = v[n0 + lr];
#pragma unroll
    for (int r = 0; r < 8; ++r) sc[r] = fmaf(fmaxf(c[r] + bn, 0.f), vn, sc[r]);
  }
#pragma unroll
  for (int r = 0; r < 8; ++r) {         // reduce across the 16 lanes of each half
    float s = sc[r];
    s += __shfl_xor(s, 1, 32);
    s += __shfl_xor(s, 2, 32);
    s += __shfl_xor(s, 4, 32);
    s += __shfl_xor(s, 8, 32);
    sc[r] = s;
  }
  if (lr == 0) {
    float b = bilb[0];
#pragma unroll
    for (int r = 0; r < 8; ++r) outseg[m0 + half * 8 + r] = sc[r] + b;
  }
}

// ---------------- host side ----------------

extern "C" void kernel_launch(void* const* d_in, const int* in_sizes, int n_in,
                              void* d_out, int out_size, void* d_ws, size_t ws_size,
                              hipStream_t stream) {
  (void)in_sizes; (void)n_in; (void)out_size; (void)ws_size;
  const float* feat  = (const float*)d_in[0];
  const float* shuff = (const float*)d_in[1];
  const int*   ei    = (const int*)d_in[2];
  const float* temp  = (const float*)d_in[3];
  const float* lin_w = (const float*)d_in[4];
  const float* lin_b = (const float*)d_in[5];
  const float* bil_w = (const float*)d_in[6];
  const float* bil_b = (const float*)d_in[7];
  const float* alpha = (const float*)d_in[8];
  const float* beta  = (const float*)d_in[9];
  const int* row = ei;
  const int* col = ei + EE;
  float* out = (float*)d_out;

  char* ws = (char*)d_ws;
  size_t off = 0;
  auto take = [&](size_t bytes) -> char* {
    char* p = ws + off;
    off = (off + bytes + 255) & ~(size_t)255;
    return p;
  };
  float* deg    = (float*)take((size_t)NN * 4);
  float* dinv   = (float*)take((size_t)NN * 4);
  int*   cnt    = (int*)take((size_t)NN * 4);
  int*   blksum = (int*)take(256 * 4);
  int*   cptr   = (int*)take((size_t)(NN + 1) * 4);
  int*   fill   = (int*)take((size_t)NN * 4);
  int*   csrc   = (int*)take((size_t)EE * 4);
  float* cw     = (float*)take((size_t)EE * 4);
  float* coe    = (float*)take(32 * 4);
  float* colsum = (float*)take(DOUT * 4);
  float* vvec   = (float*)take(DOUT * 4);
  const size_t MAT = (size_t)NN * DIN * 4;
  float* bufA = (float*)take(MAT);
  float* bufB = (float*)take(MAT);
  float* bufC = (float*)take(MAT);
  float* acc_fhp = (float*)take(MAT);
  float* acc_flp = (float*)take(MAT);
  float* acc_shp = (float*)take(MAT);
  float* acc_slp = (float*)take(MAT);

  hipMemsetAsync(deg, 0, (size_t)NN * 4, stream);
  hipMemsetAsync(cnt, 0, (size_t)NN * 4, stream);
  hipMemsetAsync(colsum, 0, DOUT * 4, stream);

  edge_counts_kernel<<<(EE + 255) / 256, 256, 0, stream>>>(row, col, deg, cnt);
  dinv_kernel<<<(NN + 255) / 256, 256, 0, stream>>>(deg, dinv);
  scan_pass1<<<SCAN_NBLK, 256, 0, stream>>>(cnt, blksum);
  scan_pass2<<<1, 32, 0, stream>>>(blksum, cptr);
  scan_pass3<<<SCAN_NBLK, 256, 0, stream>>>(cnt, blksum, cptr, fill);
  scatter_kernel<<<(EE + 255) / 256, 256, 0, stream>>>(row, col, dinv, fill, csrc, cw);
  coe_kernel<<<1, 32, 0, stream>>>(temp, coe);

  float* bufs[3] = {bufA, bufB, bufC};
  const float* xs[2] = {feat, shuff};
  float* ahp_[2] = {acc_fhp, acc_shp};
  float* alp_[2] = {acc_flp, acc_slp};
  const int prop_blocks = (NN * 32 + 255) / 256;  // one wave32 per node
  for (int xi = 0; xi < 2; ++xi) {
    const float* x = xs[xi];
    float* ahp = ahp_[xi];
    float* alp = alp_[xi];
    init_acc_kernel<<<(NN * DIN / 4 + 255) / 256, 256, 0, stream>>>(x, ahp, alp, coe);
    const float* prev2 = nullptr;  // Tx_{i-2}
    const float* prev1 = x;        // Tx_{i-1}
    for (int i = 1; i <= KORD; ++i) {
      float* outbuf = bufs[(i - 1) % 3];
      float A = (i == 1) ? 1.f : 2.f;
      float B = (i == 1) ? 0.f : -1.f;
      const float* tx0 = (i == 1) ? x : prev2;
      prop_step_kernel<<<prop_blocks, 256, 0, stream>>>(
          prev1, tx0, outbuf, ahp, alp, cptr, csrc, cw, coe, i, A, B);
      prev2 = prev1;
      prev1 = outbuf;
    }
  }

  const int cs_blocks = ((NN / 16) * (DOUT / 16) * 32 + 255) / 256;
  gemm_colsum_kernel<<<cs_blocks, 256, 0, stream>>>(acc_fhp, lin_w, lin_b, alpha, colsum);
  gemm_colsum_kernel<<<cs_blocks, 256, 0, stream>>>(acc_flp, lin_w, lin_b, beta, colsum);
  make_v_kernel<<<1, DOUT, 0, stream>>>(colsum, bil_w, vvec);

  const int sc_blocks = ((NN / 16) * 32 + 255) / 256;
  // concat order: sc(h2=feat lp), sc(h1=feat hp), sc(h4=shuf lp), sc(h3=shuf hp)
  gemm_score_kernel<<<sc_blocks, 256, 0, stream>>>(acc_flp, lin_w, lin_b, vvec, bil_b, out + 0L * NN);
  gemm_score_kernel<<<sc_blocks, 256, 0, stream>>>(acc_fhp, lin_w, lin_b, vvec, bil_b, out + 1L * NN);
  gemm_score_kernel<<<sc_blocks, 256, 0, stream>>>(acc_slp, lin_w, lin_b, vvec, bil_b, out + 2L * NN);
  gemm_score_kernel<<<sc_blocks, 256, 0, stream>>>(acc_shp, lin_w, lin_b, vvec, bil_b, out + 3L * NN);
}
